// GAT_19344532702057
// MI455X (gfx1250) — matrix operand order repaired and checked by reading, not compiled, not run on hardware
//
#include <hip/hip_runtime.h>

// GAT on MI455X (gfx1250): flash-softmax fused attention + bf16 WMMA GEMMs.
// Attention B-operand tiles staged LDS-resident via GLOBAL_LOAD_ASYNC_TO_LDS
// (double-buffered, ASYNCcnt-tracked), shared by 4 waves per workgroup.

#define NNODES 4096
#define ALPHA_SLOPE 0.2f
#define NEGV -9.0e15f

typedef __attribute__((ext_vector_type(16))) __bf16 v16bf;
typedef __attribute__((ext_vector_type(8)))  float  v8f;

__device__ __forceinline__ v8f wmma_bf16(v16bf a, v16bf b, v8f c) {
  // D(f32 16x16) = A(16x32 bf16) * B(32x16 bf16) + C
  return __builtin_amdgcn_wmma_f32_16x16x32_bf16(
      /*neg_a=*/false, a, /*neg_b=*/false, b,
      /*c_mod=*/(short)0, c, /*reuse_a=*/false, /*reuse_b=*/false);
}

// Async global -> LDS copy, 16 bytes, tracked by ASYNCcnt.
__device__ __forceinline__ void async_ld_b128(unsigned lds_off, const __bf16* g) {
  asm volatile("global_load_async_to_lds_b128 %0, %1, off"
               :: "v"(lds_off), "v"(g)
               : "memory");
}

// ---------------------------------------------------------------------------
// Transpose + convert weights: W[h][K][Nc] f32 -> WT[h][Nc][K] bf16.
// ---------------------------------------------------------------------------
__global__ void gat_w_to_bt(const float* __restrict__ W, __bf16* __restrict__ WT,
                            int K, int Nc) {
  const int h = blockIdx.y;
  W  += (size_t)h * K * Nc;
  WT += (size_t)h * K * Nc;
  int idx = blockIdx.x * blockDim.x + threadIdx.x;
  if (idx < K * Nc) {
    int k = idx / Nc;
    int n = idx - k * Nc;
    WT[(size_t)n * K + k] = (__bf16)W[idx];
  }
}

// ---------------------------------------------------------------------------
// C[h][M][64] = A[M][K] (f32, shared across heads) @ B[h][K][64] (BT = bf16,
// pre-transposed to [64][K]).  One wave per 16x64 output tile.
// ---------------------------------------------------------------------------
__global__ __launch_bounds__(32) void gat_wmma_gemm(
    const float* __restrict__ A, const __bf16* __restrict__ BT,
    float* __restrict__ C, int M, int K) {
  const int h = blockIdx.y;
  BT += (size_t)h * 64 * (size_t)K;
  C  += (size_t)h * (size_t)M * 64;

  const int lane = threadIdx.x;
  const int m16  = lane & 15;   // A row / C,B column within tile
  const int hi   = lane >> 4;   // half-wave selector
  const int row  = blockIdx.x * 16 + m16;
  const float* arow = A + (size_t)row * K;

  v8f acc[4] = {};              // 16x64 f32 accumulator (4 C fragments)

  for (int k0 = 0; k0 < K; k0 += 32) {
    // A fragment (16x32 bf16, ISA 7.12.2 layout), converted f32->bf16
    v16bf a;
#pragma unroll
    for (int run = 0; run < 2; ++run) {
      const float4* xp = (const float4*)(arow + k0 + run * 16 + hi * 8);
      float4 x0 = xp[0], x1 = xp[1];
      float xv[8] = {x0.x, x0.y, x0.z, x0.w, x1.x, x1.y, x1.z, x1.w};
#pragma unroll
      for (int t = 0; t < 8; ++t) a[run * 8 + t] = (__bf16)xv[t];
    }
    // 4 B fragments (32x16 bf16): lane = column n, two 8-K runs (ISA 7.12.2 layout)
#pragma unroll
    for (int nt = 0; nt < 4; ++nt) {
      const __bf16* bp = BT + (size_t)(nt * 16 + m16) * K + k0 + hi * 8;
      v16bf b;
#pragma unroll
      for (int t = 0; t < 16; ++t) b[t] = bp[(t >> 3) * 16 + (t & 7)];
      acc[nt] = wmma_bf16(a, b, acc[nt]);
    }
  }

#pragma unroll
  for (int nt = 0; nt < 4; ++nt)
#pragma unroll
    for (int r = 0; r < 8; ++r)
      C[(size_t)(blockIdx.x * 16 + r + hi * 8) * 64 + nt * 16 + m16] = acc[nt][r];
}

// ---------------------------------------------------------------------------
// Per-head prep: src[i] = Wh[i,:]·a[:64], dst[i] = Wh[i,:]·a[64:], and
// WhT[f][i] = bf16(Wh[i][f]) for contiguous async staging in the attention.
// ---------------------------------------------------------------------------
__global__ void gat_head_prep(const float* __restrict__ Wh,
                              const float* __restrict__ avec,
                              __bf16* __restrict__ WhT,
                              float* __restrict__ src, float* __restrict__ dst,
                              int Nn) {
  const int h = blockIdx.y;
  Wh   += (size_t)h * Nn * 64;
  avec += (size_t)h * 128;
  WhT  += (size_t)h * 64 * Nn;
  src  += (size_t)h * Nn;
  dst  += (size_t)h * Nn;

  int i = blockIdx.x * blockDim.x + threadIdx.x;
  if (i >= Nn) return;
  float s = 0.f, d = 0.f;
#pragma unroll 8
  for (int f = 0; f < 64; ++f) {
    float w = Wh[(size_t)i * 64 + f];
    s += w * avec[f];
    d += w * avec[64 + f];
    WhT[(size_t)f * Nn + i] = (__bf16)w;
  }
  src[i] = s;
  dst[i] = d;
}

// Stage one 64x32 bf16 Wh^T tile (4 KB) into LDS: 256 b128 chunks, 2/thread.
__device__ __forceinline__ void stage_tile(const __bf16* __restrict__ WhT,
                                           int Nn, int j0,
                                           __bf16* bufBase, int tid) {
#pragma unroll
  for (int s = 0; s < 2; ++s) {
    const int c = tid + s * 128;          // chunk 0..255
    const int n = c >> 2;                 // Wh^T row (feature) 0..63
    const int q = c & 3;                  // 8-element chunk within the 32 cols
    unsigned lds_off = (unsigned)(unsigned long long)(bufBase + n * 32 + q * 8);
    const __bf16* g = WhT + (size_t)n * Nn + j0 + q * 8;
    async_ld_b128(lds_off, g);
  }
}

// ---------------------------------------------------------------------------
// Fused masked-softmax attention: 4 waves/block, each owning a 16-row block of
// the same head; 64x32 Wh^T tile double-buffered in LDS via async copies.
// out[16x64] = softmax(mask ? lrelu(src_i+dst_j) : NEG) @ Wh, optional ELU.
// ---------------------------------------------------------------------------
__global__ __launch_bounds__(128) void gat_flash_attn(
    const __bf16* __restrict__ WhT, const float* __restrict__ src,
    const float* __restrict__ dst, const int* __restrict__ adj,
    float* __restrict__ out, int Nn, int row_stride, int do_elu) {
  __shared__ __bf16 sB[2][64 * 32];       // double-buffered Wh^T tile (2x4 KB)

  const int h = blockIdx.y;
  WhT += (size_t)h * 64 * Nn;
  src += (size_t)h * Nn;
  dst += (size_t)h * Nn;
  const int col_off = h * 64;

  const int tid  = threadIdx.x;
  const int wv   = tid >> 5;              // wave within block -> row block
  const int lane = tid & 31;
  const int m16  = lane & 15;             // matrix row for the A(prob) fragment
  const int hi   = lane >> 4;
  const int rb   = blockIdx.x * 4 + wv;   // 16-row block index
  const int row  = rb * 16 + m16;

  const float s_i = src[row];
  float m_run = -3.0e38f;                 // running row max
  float l_run = 0.0f;                     // running row sum
  v8f acc[4] = {};

  stage_tile(WhT, Nn, 0, &sB[0][0], tid); // prologue: tile 0 -> buffer 0

  const int ntiles = Nn / 32;
  for (int jt = 0; jt < ntiles; ++jt) {
    const int cur = jt & 1;
    const int j0  = jt * 32;

    __syncthreads();                      // all waves done reading sB[1-cur]
    if (jt + 1 < ntiles) {
      stage_tile(WhT, Nn, j0 + 32, &sB[1 - cur][0], tid);
      asm volatile("s_wait_asynccnt 0x2" ::: "memory");  // tile jt landed
    } else {
      asm volatile("s_wait_asynccnt 0x0" ::: "memory");
    }
    __syncthreads();                      // tile jt visible to all waves

    // keep the adjacency stream ahead of the exp-heavy VALU work
    __builtin_prefetch(adj + (size_t)row * Nn + j0 + 64, 0, 0);

    // ---- scores for this lane's 16 columns (two contiguous 8-col runs) ----
    float ev[16];
    float tmax = -3.0e38f;
#pragma unroll
    for (int run = 0; run < 2; ++run) {
      const int jb = j0 + run * 16 + hi * 8;
      const float4* dp = (const float4*)(dst + jb);
      float4 d0 = dp[0], d1 = dp[1];
      const int4* ap = (const int4*)(adj + (size_t)row * Nn + jb);
      int4 A0 = ap[0], A1 = ap[1];
      float dv[8] = {d0.x, d0.y, d0.z, d0.w, d1.x, d1.y, d1.z, d1.w};
      int   av[8] = {A0.x, A0.y, A0.z, A0.w, A1.x, A1.y, A1.z, A1.w};
#pragma unroll
      for (int t = 0; t < 8; ++t) {
        float e = s_i + dv[t];
        e = (e > 0.0f) ? e : ALPHA_SLOPE * e;   // LeakyReLU
        e = av[t] ? e : NEGV;                   // adjacency mask
        ev[run * 8 + t] = e;
        tmax = fmaxf(tmax, e);
      }
    }
    // both half-wave lanes of a row agree on row statistics
    tmax = fmaxf(tmax, __shfl_xor(tmax, 16, 32));
    const float m_new = fmaxf(m_run, tmax);
    const float scl   = __expf(m_run - m_new);

    float psum = 0.0f;
    v16bf pa;                             // probability tile in A-frag order
#pragma unroll
    for (int t = 0; t < 16; ++t) {
      float p = __expf(ev[t] - m_new);    // masked -> exp(-huge) = 0
      psum += p;
      pa[t] = (__bf16)p;
    }
    psum += __shfl_xor(psum, 16, 32);
    l_run = l_run * scl + psum;
    m_run = m_new;

    // broadcast per-row rescale to C-fragment lane mapping (lane r holds it)
    float rs[8];
#pragma unroll
    for (int r = 0; r < 8; ++r) rs[r] = __shfl(scl, r + hi * 8, 32);
#pragma unroll
    for (int r = 0; r < 8; ++r) {
      acc[0][r] *= rs[r]; acc[1][r] *= rs[r];
      acc[2][r] *= rs[r]; acc[3][r] *= rs[r];
    }

    // ---- P(16x32) @ Wh(32x64): 4 WMMAs, B fragments from LDS-resident tile
#pragma unroll
    for (int nt = 0; nt < 4; ++nt) {
      const __bf16* bp = &sB[cur][(nt * 16 + m16) * 32 + hi * 8];
      v16bf b;
#pragma unroll
      for (int t = 0; t < 16; ++t) b[t] = bp[(t >> 3) * 16 + (t & 7)];
      acc[nt] = wmma_bf16(pa, b, acc[nt]);
    }
  }

  // epilogue: normalize by row sum, optional ELU, store concat layout
#pragma unroll
  for (int r = 0; r < 8; ++r) {
    const float lr = __shfl(l_run, r + hi * 8, 32);
    const int ro = rb * 16 + r + hi * 8;
#pragma unroll
    for (int nt = 0; nt < 4; ++nt) {
      float v = acc[nt][r] / lr;
      if (do_elu) v = (v > 0.0f) ? v : (__expf(v) - 1.0f);
      out[(size_t)ro * row_stride + col_off + nt * 16 + m16] = v;
    }
  }
}

// ---------------------------------------------------------------------------
extern "C" void kernel_launch(void* const* d_in, const int* in_sizes, int n_in,
                              void* d_out, int out_size, void* d_ws, size_t ws_size,
                              hipStream_t stream) {
  const float* x    = (const float*)d_in[0];   // [4096,512]
  const int*   adj  = (const int*)d_in[1];     // [4096,4096]
  const float* W0   = (const float*)d_in[2];   // [4,512,64]
  const float* a0   = (const float*)d_in[3];   // [4,128,1]
  const float* W1   = (const float*)d_in[4];   // [4,256,64]
  const float* a1   = (const float*)d_in[5];   // [4,128,1]
  const float* Wout = (const float*)d_in[6];   // [256,64]
  const float* aout = (const float*)d_in[7];   // [128,1]
  float* out = (float*)d_out;                  // [4096,64]

  // workspace carve-out (256B aligned slices)
  char* ws = (char*)d_ws;
  auto carve = [&](size_t bytes) -> char* {
    char* p = ws;
    ws += (bytes + 255) & ~(size_t)255;
    return p;
  };
  __bf16* WT0  = (__bf16*)carve((size_t)4 * 64 * 512 * 2);
  __bf16* WT1  = (__bf16*)carve((size_t)4 * 64 * 256 * 2);
  __bf16* WTo  = (__bf16*)carve((size_t)64 * 256 * 2);
  float*  Wh   = (float*)carve((size_t)4 * NNODES * 64 * 4);
  __bf16* WhT  = (__bf16*)carve((size_t)4 * 64 * NNODES * 2);
  float*  srcb = (float*)carve((size_t)4 * NNODES * 4);
  float*  dstb = (float*)carve((size_t)4 * NNODES * 4);
  float*  h0   = (float*)carve((size_t)NNODES * 256 * 4);
  float*  h1   = (float*)carve((size_t)NNODES * 256 * 4);

  const dim3 b32(32, 1, 1);
  const dim3 b128(128, 1, 1);
  const int rowBlocks  = NNODES / 16;          // 256 (GEMM grid)
  const int attnBlocks = NNODES / 64;          // 64  (4 row-blocks per WG)

  // ---------------- layer 0: x[4096,512] -> h0[4096,256] ----------------
  gat_w_to_bt<<<dim3((512 * 64 + 255) / 256, 4, 1), 256, 0, stream>>>(W0, WT0, 512, 64);
  gat_wmma_gemm<<<dim3(rowBlocks, 4, 1), b32, 0, stream>>>(x, WT0, Wh, NNODES, 512);
  gat_head_prep<<<dim3(NNODES / 128, 4, 1), 128, 0, stream>>>(Wh, a0, WhT, srcb, dstb, NNODES);
  gat_flash_attn<<<dim3(attnBlocks, 4, 1), b128, 0, stream>>>(WhT, srcb, dstb, adj, h0,
                                                              NNODES, 256, /*elu=*/1);

  // ---------------- layer 1: h0[4096,256] -> h1[4096,256] ----------------
  gat_w_to_bt<<<dim3((256 * 64 + 255) / 256, 4, 1), 256, 0, stream>>>(W1, WT1, 256, 64);
  gat_wmma_gemm<<<dim3(rowBlocks, 4, 1), b32, 0, stream>>>(h0, WT1, Wh, NNODES, 256);
  gat_head_prep<<<dim3(NNODES / 128, 4, 1), 128, 0, stream>>>(Wh, a1, WhT, srcb, dstb, NNODES);
  gat_flash_attn<<<dim3(attnBlocks, 4, 1), b128, 0, stream>>>(WhT, srcb, dstb, adj, h1,
                                                              NNODES, 256, /*elu=*/1);

  // ---------------- output layer: h1[4096,256] -> out[4096,64] -----------
  gat_w_to_bt<<<dim3((256 * 64 + 255) / 256, 1, 1), 256, 0, stream>>>(Wout, WTo, 256, 64);
  gat_wmma_gemm<<<dim3(rowBlocks, 1, 1), b32, 0, stream>>>(h1, WTo, Wh, NNODES, 256);
  gat_head_prep<<<dim3(NNODES / 128, 1, 1), 128, 0, stream>>>(Wh, aout, WhT, srcb, dstb, NNODES);
  gat_flash_attn<<<dim3(attnBlocks, 1, 1), b128, 0, stream>>>(WhT, srcb, dstb, adj, out,
                                                              NNODES, 64, /*elu=*/0);
}
